// AtlasMAGBlock_80616536146619
// MI455X (gfx1250) — matrix operand order repaired and checked by reading, not compiled, not run on hardware
//
#include <hip/hip_runtime.h>
#include <hip/hip_bf16.h>
#include <stdint.h>
#include <stddef.h>

// ---------------- problem constants ----------------
#define Bq    2
#define Sq    2048
#define Dq    1024
#define Hq    16
#define HDq   64
#define FPOLY (HDq + HDq*HDq)   // 4160
#define MHID  (2*Dq)            // 2048
#define FFNH  2730
#define FFNHP 2752              // padded leading dim (multiple of 64, 16B aligned f16)
#define GGH   64
#define EPSq  1e-6f
#define OMW   64
#define OMDEC 0.95f
#define TTLA  0.999f
#define TTLE  0.01f
#define NSA   3.4445f
#define NSB  (-4.775f)
#define NSC   2.0315f

typedef __attribute__((ext_vector_type(16))) _Float16 v16h;
typedef __attribute__((ext_vector_type(8)))  _Float16 v8h;
typedef __attribute__((ext_vector_type(8)))  float    v8f;
typedef __attribute__((ext_vector_type(4)))  unsigned int v4u;
typedef __attribute__((ext_vector_type(8)))  int      v8i;
typedef __attribute__((ext_vector_type(4)))  int      v4i;

#if __has_builtin(__builtin_amdgcn_tensor_load_to_lds)
#define HAVE_TDM 1
#else
#define HAVE_TDM 0
#endif

#if HAVE_TDM
// One-shot 2D TDM tile load: rows x 32 halves, global row stride ld_elems
// halves, into LDS rows padded 64B data + 16B pad (stride 40 halves).
// D# layout per CDNA5 ISA (group0 128b, group1 256b).
__device__ __forceinline__ void tdm_load_tile_2d(
    const _Float16* gsrc, unsigned lds_byte_addr, int rows, int ld_elems)
{
    unsigned long long ga = (unsigned long long)(uintptr_t)gsrc;
    v4u g0;
    g0[0] = 1u;                                          // count=1, user mode
    g0[1] = lds_byte_addr;                               // lds_addr
    g0[2] = (unsigned)(ga & 0xffffffffu);                // global_addr[31:0]
    g0[3] = (unsigned)((ga >> 32) & 0x01ffffffu)         // global_addr[56:32]
          | (2u << 30);                                  // type=2 ("image")

    const unsigned td0 = 32u;                            // tensor_dim0 (= tile cols)
    const unsigned td1 = (unsigned)rows;                 // tensor_dim1
    const unsigned long long st0 = (unsigned long long)(unsigned)ld_elems;
    v8i g1;
    g1[0] = (int)((1u << 16)        // data_size = 2B
                | (1u << 20)        // pad_enable
                | (3u << 22)        // pad_interval: 64B of data
                | (3u << 25));      // pad_amount: 4 DWORDs (16B)
    g1[1] = (int)((td0 & 0xffffu) << 16);                          // tensor_dim0 lo
    g1[2] = (int)((td0 >> 16) | ((td1 & 0xffffu) << 16));          // dim0 hi | dim1 lo
    g1[3] = (int)((td1 >> 16) | (32u << 16));                      // dim1 hi | tile_dim0=32
    g1[4] = (int)(unsigned)rows;                                   // tile_dim1 (tile_dim2=0)
    g1[5] = (int)(st0 & 0xffffffffu);                              // tensor_dim0_stride lo
    g1[6] = (int)((st0 >> 32) & 0xffffu);                          // stride hi | dim1_stride=0
    g1[7] = 0;
    v4i gz = { 0, 0, 0, 0 };
#if __clang_major__ >= 23
    v8i gz8 = { 0, 0, 0, 0, 0, 0, 0, 0 };
    __builtin_amdgcn_tensor_load_to_lds(g0, g1, gz, gz, gz8, 0);
#else
    __builtin_amdgcn_tensor_load_to_lds(g0, g1, gz, gz, 0);
#endif
}
#endif

// =====================================================================
// Tiled WMMA GEMM on f16 operands, f32 accumulate:
//   C[M,N] = op(A)[M,K] @ op(B)[K,N]
//   transA: A[m,k] = Aptr[k*lda + m] else Aptr[m*lda + k]
//   transB: B[k,n] = Bptr[n*ldb + k] else Bptr[k*ldb + n]
//   mode 1: silu epilogue.  Eadd: C += alpha_add * Eadd[m*ldc+n].
// Block 128 thr = 4 waves; tile 64x64; k-step 32. Straight-copy stages go
// through the Tensor Data Mover; transpose stages go through VGPRs.
// =====================================================================
#define TM 64
#define TN 64
#define TK 32

__global__ __launch_bounds__(128)
void k_gemm(const _Float16* __restrict__ A, const _Float16* __restrict__ Bm,
            float* __restrict__ C, const float* __restrict__ Eadd,
            int M, int N, int K, int lda, int ldb, int ldc,
            int transA, int transB, int mode, float alpha_add)
{
    __shared__ _Float16 As [TM][TK + 8];   // [m][k]
    __shared__ _Float16 Bst[TN][TK + 8];   // [n][k] (transposed stage)

    const int tid  = threadIdx.x;
    const int wave = tid >> 5;
    const int lane = tid & 31;
    const int bm = blockIdx.x * TM;
    const int bn = blockIdx.y * TN;
    const int wm = (wave >> 1) * 32;
    const int wn = (wave & 1)  * 32;
    const int mlane = lane & 15;
    const int hl    = lane >> 4;

    v8f acc[2][2] = {};

    for (int k0 = 0; k0 < K; k0 += TK) {
        const bool fullK = (k0 + TK <= K);
        const bool fullA = fullK && (bm + TM <= M);
        const bool fullB = fullK && (bn + TN <= N);

#if HAVE_TDM
        const bool tdmA = (transA == 0) && fullA;
        const bool tdmB = (transB != 0) && fullB;
        if ((tdmA || tdmB) && wave == 0) {
            if (tdmA) tdm_load_tile_2d(A + (size_t)bm * lda + k0,
                                       (unsigned)(uintptr_t)&As[0][0], TM, lda);
            if (tdmB) tdm_load_tile_2d(Bm + (size_t)bn * ldb + k0,
                                       (unsigned)(uintptr_t)&Bst[0][0], TN, ldb);
            __builtin_amdgcn_s_wait_tensorcnt(0);
        }
#else
        const bool tdmA = false, tdmB = false;
#endif

        if ((tid & 63) == 0 && k0 + TK < K) {   // hint next k-tile
            __builtin_prefetch(A  + (transA ? (size_t)(k0+TK)*lda + bm
                                            : (size_t)bm*lda + k0 + TK), 0, 1);
            __builtin_prefetch(Bm + (transB ? (size_t)bn*ldb + k0 + TK
                                            : (size_t)(k0+TK)*ldb + bn), 0, 1);
        }

        // ---------------- stage A (if not DMA'd) ----------------
        if (!tdmA) {
            if (!transA) {
                if (fullA) {
                    for (int ch = tid; ch < TM * (TK/8); ch += 128) {
                        int r = ch >> 2, c8 = (ch & 3) << 3;
                        v8h v = *(const v8h*)(A + (size_t)(bm + r) * lda + k0 + c8);
                        *(v8h*)&As[r][c8] = v;
                    }
                } else {
                    for (int idx = tid; idx < TM * TK; idx += 128) {
                        int r = idx >> 5, c = idx & 31;
                        int m = bm + r, k = k0 + c;
                        As[r][c] = (m < M && k < K) ? A[(size_t)m*lda + k] : (_Float16)0.f;
                    }
                }
            } else {
                if (fullA) {
                    for (int ch = tid; ch < TK * (TM/8); ch += 128) {
                        int c = ch >> 3, r8 = (ch & 7) << 3;        // c = k idx
                        v8h v = *(const v8h*)(A + (size_t)(k0 + c) * lda + bm + r8);
                        #pragma unroll
                        for (int i = 0; i < 8; ++i) As[r8 + i][c] = v[i];
                    }
                } else {
                    for (int idx = tid; idx < TM * TK; idx += 128) {
                        int r = idx >> 5, c = idx & 31;
                        int m = bm + r, k = k0 + c;
                        As[r][c] = (m < M && k < K) ? A[(size_t)k*lda + m] : (_Float16)0.f;
                    }
                }
            }
        }
        // ---------------- stage B into Bst[n][k] (if not DMA'd) ----------------
        if (!tdmB) {
            if (!transB) {
                if (fullB) {
                    for (int ch = tid; ch < TK * (TN/8); ch += 128) {
                        int r = ch >> 3, n8 = (ch & 7) << 3;        // r = k idx
                        v8h v = *(const v8h*)(Bm + (size_t)(k0 + r) * ldb + bn + n8);
                        #pragma unroll
                        for (int i = 0; i < 8; ++i) Bst[n8 + i][r] = v[i];
                    }
                } else {
                    for (int idx = tid; idx < TM * TK; idx += 128) {
                        int r = idx >> 5, c = idx & 31;             // r = n, c = k
                        int n = bn + r, k = k0 + c;
                        Bst[r][c] = (k < K && n < N) ? Bm[(size_t)k*ldb + n] : (_Float16)0.f;
                    }
                }
            } else {
                if (fullB) {
                    for (int ch = tid; ch < TN * (TK/8); ch += 128) {
                        int r = ch >> 2, c8 = (ch & 3) << 3;        // r = n
                        v8h v = *(const v8h*)(Bm + (size_t)(bn + r) * ldb + k0 + c8);
                        *(v8h*)&Bst[r][c8] = v;
                    }
                } else {
                    for (int idx = tid; idx < TM * TK; idx += 128) {
                        int r = idx >> 5, c = idx & 31;
                        int n = bn + r, k = k0 + c;
                        Bst[r][c] = (k < K && n < N) ? Bm[(size_t)n*ldb + k] : (_Float16)0.f;
                    }
                }
            }
        }
        __syncthreads();

        // ---------------- fragments (contiguous 16B LDS loads) ----------------
        v16h af[2], bf[2];
        #pragma unroll
        for (int i = 0; i < 2; ++i) {
            const _Float16* pr = &As[wm + 16*i + mlane][8 * hl];
            v8h lo = *(const v8h*)pr;
            v8h hi = *(const v8h*)(pr + 16);
            af[i] = __builtin_shufflevector(lo, hi, 0,1,2,3,4,5,6,7,8,9,10,11,12,13,14,15);
        }
        #pragma unroll
        for (int j = 0; j < 2; ++j) {
            const _Float16* pr = &Bst[wn + 16*j + mlane][8 * hl];
            v8h lo = *(const v8h*)pr;
            v8h hi = *(const v8h*)(pr + 16);
            bf[j] = __builtin_shufflevector(lo, hi, 0,1,2,3,4,5,6,7,8,9,10,11,12,13,14,15);
        }
        #pragma unroll
        for (int i = 0; i < 2; ++i)
            #pragma unroll
            for (int j = 0; j < 2; ++j)
                acc[i][j] = __builtin_amdgcn_wmma_f32_16x16x32_f16(
                    false, af[i], false, bf[j], (short)0, acc[i][j], false, false);
        __syncthreads();
    }

    // ---------------- epilogue ----------------
    #pragma unroll
    for (int i = 0; i < 2; ++i)
        #pragma unroll
        for (int j = 0; j < 2; ++j)
            #pragma unroll
            for (int r = 0; r < 8; ++r) {
                int m = bm + wm + 16*i + 8*hl + r;
                int n = bn + wn + 16*j + mlane;
                if (m < M && n < N) {
                    float v = acc[i][j][r];
                    if (Eadd) v += alpha_add * Eadd[(size_t)m * ldc + n];
                    if (mode == 1) v = v / (1.f + __expf(-v));
                    C[(size_t)m * ldc + n] = v;
                }
            }
}

static inline void gemm(hipStream_t s, const _Float16* A, const _Float16* B, float* C,
                        int M, int N, int K, int lda, int ldb, int ldc,
                        int tA, int tB, int mode = 0,
                        const float* Eadd = nullptr, float aadd = 0.f)
{
    dim3 g((unsigned)((M + TM - 1) / TM), (unsigned)((N + TN - 1) / TN));
    k_gemm<<<g, dim3(128), 0, s>>>(A, B, C, Eadd, M, N, K, lda, ldb, ldc, tA, tB, mode, aadd);
}

// =====================================================================
// Flash attention (causal). Q16,K16: (B,H,S,HD) f16; Vt16: (B,H,HD,S) f16.
// Out16: (B,S,D) f16 (feeds the wo GEMM). 1 wave = 16 queries.
// =====================================================================
__global__ __launch_bounds__(128)
void k_flash(const _Float16* __restrict__ Q16, const _Float16* __restrict__ K16,
             const _Float16* __restrict__ Vt16, _Float16* __restrict__ Out16)
{
    __shared__ _Float16 p_lds[4][16][TK + 8];

    const int tid = threadIdx.x, wave = tid >> 5, lane = tid & 31;
    const int mlane = lane & 15, hl = lane >> 4;

    const int chunks = Sq / 64;
    const int blk   = blockIdx.x;
    const int chunk = blk % chunks;
    const int bh    = blk / chunks;
    const int b  = bh / Hq;
    const int hh = bh % Hq;

    const _Float16* Qh = Q16 + (size_t)bh * Sq * HDq;
    const _Float16* Kh = K16 + (size_t)bh * Sq * HDq;
    const _Float16* Vh = Vt16 + (size_t)bh * HDq * Sq;

    const int qbase = chunk * 64 + wave * 16;

    v16h qa[2];
    #pragma unroll
    for (int i = 0; i < 2; ++i) {
        const _Float16* pr = Qh + (size_t)(qbase + mlane) * HDq + 32*i + 8*hl;
        v8h lo = *(const v8h*)pr;
        v8h hi = *(const v8h*)(pr + 16);
        qa[i] = __builtin_shufflevector(lo, hi, 0,1,2,3,4,5,6,7,8,9,10,11,12,13,14,15);
    }

    float mrow[8], lrow[8];
    v8f o[4] = {};
    #pragma unroll
    for (int r = 0; r < 8; ++r) { mrow[r] = -1e30f; lrow[r] = 0.f; }

    const int tend = qbase + 16;
    for (int t0 = 0; t0 < tend; t0 += 32) {
        v8f s[2] = {};
        #pragma unroll
        for (int half = 0; half < 2; ++half)
            #pragma unroll
            for (int dh = 0; dh < 2; ++dh) {
                const _Float16* pr = Kh + (size_t)(t0 + 16*half + mlane) * HDq + 32*dh + 8*hl;
                v8h lo = *(const v8h*)pr;
                v8h hi = *(const v8h*)(pr + 16);
                v16h kb = __builtin_shufflevector(lo, hi, 0,1,2,3,4,5,6,7,8,9,10,11,12,13,14,15);
                s[half] = __builtin_amdgcn_wmma_f32_16x16x32_f16(
                    false, qa[dh], false, kb, (short)0, s[half], false, false);
            }

        #pragma unroll
        for (int r = 0; r < 8; ++r) {
            int qg = qbase + 8*hl + r;
            float s0 = s[0][r] * 0.125f;
            float s1 = s[1][r] * 0.125f;
            if (t0 + mlane      > qg) s0 = -1e30f;
            if (t0 + 16 + mlane > qg) s1 = -1e30f;
            float mx = fmaxf(s0, s1);
            for (int off = 1; off < 16; off <<= 1)
                mx = fmaxf(mx, __shfl_xor(mx, off, 16));
            float mnew  = fmaxf(mrow[r], mx);
            float alpha = __expf(mrow[r] - mnew);
            float p0 = __expf(s0 - mnew);
            float p1 = __expf(s1 - mnew);
            float rs = p0 + p1;
            for (int off = 1; off < 16; off <<= 1)
                rs += __shfl_xor(rs, off, 16);
            lrow[r] = lrow[r] * alpha + rs;
            mrow[r] = mnew;
            #pragma unroll
            for (int j = 0; j < 4; ++j) o[j][r] *= alpha;
            p_lds[wave][8*hl + r][mlane]      = (_Float16)p0;
            p_lds[wave][8*hl + r][16 + mlane] = (_Float16)p1;
        }

        v16h pa;
        {
            const _Float16* pr = &p_lds[wave][mlane][8 * hl];
            v8h lo = *(const v8h*)pr;
            v8h hi = *(const v8h*)(pr + 16);
            pa = __builtin_shufflevector(lo, hi, 0,1,2,3,4,5,6,7,8,9,10,11,12,13,14,15);
        }
        #pragma unroll
        for (int j = 0; j < 4; ++j) {
            const _Float16* pr = Vh + (size_t)(16*j + mlane) * Sq + t0 + 8*hl;
            v8h lo = *(const v8h*)pr;
            v8h hi = *(const v8h*)(pr + 16);
            v16h vb = __builtin_shufflevector(lo, hi, 0,1,2,3,4,5,6,7,8,9,10,11,12,13,14,15);
            o[j] = __builtin_amdgcn_wmma_f32_16x16x32_f16(
                false, pa, false, vb, (short)0, o[j], false, false);
        }
    }

    #pragma unroll
    for (int j = 0; j < 4; ++j)
        #pragma unroll
        for (int r = 0; r < 8; ++r) {
            int qg = qbase + 8*hl + r;
            Out16[((size_t)b * Sq + qg) * Dq + hh * HDq + 16*j + mlane] =
                (_Float16)(o[j][r] / lrow[r]);
        }
}

// =====================================================================
// Elementwise / reduction kernels
// =====================================================================
__global__ __launch_bounds__(256)
void k_rmsnorm(const float* __restrict__ x, const float* __restrict__ w,
               _Float16* __restrict__ o)
{
    int row = blockIdx.x;
    const float* xr = x + (size_t)row * Dq;
    _Float16* orow = o + (size_t)row * Dq;
    __shared__ float red[256];
    float a = 0.f;
    for (int i = threadIdx.x; i < Dq; i += 256) { float v = xr[i]; a += v * v; }
    red[threadIdx.x] = a; __syncthreads();
    for (int st = 128; st > 0; st >>= 1) {
        if ((int)threadIdx.x < st) red[threadIdx.x] += red[threadIdx.x + st];
        __syncthreads();
    }
    float sc = rsqrtf(red[0] / (float)Dq + EPSq);
    for (int i = threadIdx.x; i < Dq; i += 256) orow[i] = (_Float16)(xr[i] * sc * w[i]);
}

__global__ __launch_bounds__(64)
void k_qkv_heads(const float* __restrict__ qkv, const float* __restrict__ qw,
                 const float* __restrict__ kw, float* __restrict__ Qo,
                 float* __restrict__ Ko, _Float16* __restrict__ Q16,
                 _Float16* __restrict__ K16, _Float16* __restrict__ Vt16,
                 float* __restrict__ Vflat)
{
    int blk = blockIdx.x;
    int h = blk % Hq; int bs = blk / Hq;
    int s = bs % Sq;  int b = bs / Sq;
    int i = threadIdx.x;

    const float* row = qkv + (size_t)bs * 3 * Dq;
    float qv = row[h * HDq + i];
    float kv = row[Dq + h * HDq + i];
    float vv = row[2 * Dq + h * HDq + i];

    __shared__ float red[64];
    __shared__ float sq[64], skk[64];

    red[i] = qv * qv; __syncthreads();
    for (int st = 32; st > 0; st >>= 1) { if (i < st) red[i] += red[i + st]; __syncthreads(); }
    float qss = rsqrtf(red[0] / (float)HDq + EPSq); __syncthreads();
    red[i] = kv * kv; __syncthreads();
    for (int st = 32; st > 0; st >>= 1) { if (i < st) red[i] += red[i + st]; __syncthreads(); }
    float kss = rsqrtf(red[0] / (float)HDq + EPSq); __syncthreads();

    sq[i]  = qv * qss * qw[i];
    skk[i] = kv * kss * kw[i];
    __syncthreads();

    const int half = HDq / 2;
    float oq, ok;
    if (i < half) {
        float inv = __powf(10000.f, -(float)i / (float)half);
        float fr = (float)s * inv, c = __cosf(fr), sn = __sinf(fr);
        oq = sq[i] * c - sq[i + half] * sn;
        ok = skk[i] * c - skk[i + half] * sn;
    } else {
        int i0 = i - half;
        float inv = __powf(10000.f, -(float)i0 / (float)half);
        float fr = (float)s * inv, c = __cosf(fr), sn = __sinf(fr);
        oq = sq[i0] * sn + sq[i] * c;
        ok = skk[i0] * sn + skk[i] * c;
    }
    int bh = b * Hq + h;
    size_t ix = ((size_t)bh * Sq + s) * HDq + i;
    Qo[ix] = oq; Ko[ix] = ok;
    Q16[ix] = (_Float16)oq; K16[ix] = (_Float16)ok;
    Vt16[((size_t)bh * HDq + i) * Sq + s] = (_Float16)vv;
    Vflat[(size_t)bs * Dq + h * HDq + i] = vv;
}

__global__ __launch_bounds__(64)
void k_gamma(const float* __restrict__ gm, const float* __restrict__ g2w,
             float* __restrict__ gamma, float* __restrict__ womega)
{
    int t = blockIdx.x, i = threadIdx.x;
    __shared__ float red[64];
    red[i] = gm[(size_t)t * GGH + i] * g2w[i];
    __syncthreads();
    for (int st = 32; st > 0; st >>= 1) { if (i < st) red[i] += red[i + st]; __syncthreads(); }
    if (i == 0) {
        float g = 1.f / (1.f + __expf(-red[0]));
        gamma[t] = g;
        int s = t % Sq;
        float dp = (s >= Sq - OMW) ? __powf(OMDEC, (float)(Sq - 1 - s)) : 0.f;
        womega[t] = g * dp;
    }
}

__global__ __launch_bounds__(64)
void k_kcm_qmem(const float* __restrict__ Ko, const float* __restrict__ Qo,
                const float* __restrict__ gamma, _Float16* __restrict__ qmem16)
{
    int bh = blockIdx.x; int b = bh / Hq; int h = bh % Hq;
    int d = threadIdx.x;
    const float* kp = Ko + (size_t)bh * Sq * HDq;
    const float* qp = Qo + (size_t)bh * Sq * HDq;
    float acc = 0.f;
    for (int s = 0; s < Sq; ++s) {
        acc += kp[(size_t)s * HDq + d];
        float kcm = acc / (float)(s + 1);
        float g = gamma[b * Sq + s];
        qmem16[((size_t)b * Sq + s) * Dq + h * HDq + d] =
            (_Float16)(g * qp[(size_t)s * HDq + d] + (1.f - g) * kcm);
    }
}

__global__ __launch_bounds__(256)
void k_phi2(const float* __restrict__ z, _Float16* __restrict__ feat16)
{
    int t = blockIdx.x;
    const float* zp = z + (size_t)t * HDq;
    _Float16* fp = feat16 + (size_t)t * FPOLY;
    __shared__ float zs[HDq];
    if (threadIdx.x < HDq) zs[threadIdx.x] = zp[threadIdx.x];
    __syncthreads();
    for (int idx = threadIdx.x; idx < FPOLY; idx += 256) {
        float v;
        if (idx < HDq) v = zs[idx];
        else { int o = idx - HDq; v = zs[o >> 6] * zs[o & 63] * 0.125f; }
        fp[idx] = (_Float16)v;
    }
}

__global__ __launch_bounds__(64)
void k_dz(const float* __restrict__ dfeat, const float* __restrict__ z,
          float* __restrict__ dz)
{
    int t = blockIdx.x, i = threadIdx.x;
    __shared__ float zs[HDq];
    zs[i] = z[(size_t)t * HDq + i];
    __syncthreads();
    const float* df = dfeat + (size_t)t * FPOLY;
    float acc = 0.f;
    for (int j = 0; j < HDq; ++j)
        acc += (df[HDq + i * HDq + j] + df[HDq + j * HDq + i]) * zs[j];
    dz[(size_t)t * HDq + i] = df[i] + acc * 0.125f;
}

__global__ void k_silu_h(const float* __restrict__ a, _Float16* __restrict__ o, size_t n)
{
    for (size_t i = blockIdx.x * (size_t)blockDim.x + threadIdx.x; i < n;
         i += (size_t)gridDim.x * blockDim.x) {
        float u = a[i]; o[i] = (_Float16)(u / (1.f + __expf(-u)));
    }
}

__global__ void k_silu_bwd(const float* __restrict__ dh, const float* __restrict__ u,
                           _Float16* __restrict__ du16, size_t n)
{
    for (size_t i = blockIdx.x * (size_t)blockDim.x + threadIdx.x; i < n;
         i += (size_t)gridDim.x * blockDim.x) {
        float uu = u[i];
        float sg = 1.f / (1.f + __expf(-uu));
        du16[i] = (_Float16)(dh[i] * sg * (1.f + uu * (1.f - sg)));
    }
}

__global__ void k_dpred(const float* __restrict__ pred, const float* __restrict__ vf,
                        const float* __restrict__ wo, const float* __restrict__ sumw,
                        _Float16* __restrict__ dp16, size_t n)
{
    float inv = 2.f / (sumw[0] + 1e-8f);
    for (size_t i = blockIdx.x * (size_t)blockDim.x + threadIdx.x; i < n;
         i += (size_t)gridDim.x * blockDim.x) {
        size_t t = i / Dq;
        dp16[i] = (_Float16)(inv * wo[t] * (pred[i] - vf[i]));
    }
}

__global__ void k_reduce_partial(const float* __restrict__ v, size_t n,
                                 float* __restrict__ partial, int sq)
{
    __shared__ float red[256];
    float a = 0.f;
    for (size_t i = blockIdx.x * (size_t)blockDim.x + threadIdx.x; i < n;
         i += (size_t)gridDim.x * blockDim.x) {
        float x = v[i]; a += sq ? x * x : x;
    }
    red[threadIdx.x] = a; __syncthreads();
    for (int st = 128; st > 0; st >>= 1) {
        if ((int)threadIdx.x < st) red[threadIdx.x] += red[threadIdx.x + st];
        __syncthreads();
    }
    if (threadIdx.x == 0) partial[blockIdx.x] = red[0];
}

__global__ void k_reduce_final(const float* __restrict__ partial, int nb,
                               float* __restrict__ out)
{
    __shared__ float red[256];
    float a = 0.f;
    for (int i = threadIdx.x; i < nb; i += 256) a += partial[i];
    red[threadIdx.x] = a; __syncthreads();
    for (int st = 128; st > 0; st >>= 1) {
        if ((int)threadIdx.x < st) red[threadIdx.x] += red[threadIdx.x + st];
        __syncthreads();
    }
    if (threadIdx.x == 0) out[0] = red[0];
}

__global__ void k_transpose_scale(const float* __restrict__ G, float* __restrict__ X,
                                  int r, int c, const float* __restrict__ ss)
{
    float sc = 1.f / (sqrtf(ss[0]) + 1e-7f);
    size_t n = (size_t)r * c;
    for (size_t i = blockIdx.x * (size_t)blockDim.x + threadIdx.x; i < n;
         i += (size_t)gridDim.x * blockDim.x) {
        int rr = (int)(i / c), cc = (int)(i % c);
        X[(size_t)cc * r + rr] = G[i] * sc;
    }
}

__global__ void k_ns_combine(const float* __restrict__ A, const float* __restrict__ A2,
                             float* __restrict__ T, size_t n)
{
    for (size_t i = blockIdx.x * (size_t)blockDim.x + threadIdx.x; i < n;
         i += (size_t)gridDim.x * blockDim.x)
        T[i] = NSB * A[i] + NSC * A2[i];
}

__global__ void k_cvtf(const float* __restrict__ s, _Float16* __restrict__ d, size_t n)
{
    for (size_t i = blockIdx.x * (size_t)blockDim.x + threadIdx.x; i < n;
         i += (size_t)gridDim.x * blockDim.x)
        d[i] = (_Float16)s[i];
}

__global__ void k_cvtp(const float* __restrict__ s, _Float16* __restrict__ d,
                       int R, int Cs, int ldd)
{
    size_t n = (size_t)R * ldd;
    for (size_t i = blockIdx.x * (size_t)blockDim.x + threadIdx.x; i < n;
         i += (size_t)gridDim.x * blockDim.x) {
        int r = (int)(i / ldd), c = (int)(i % ldd);
        d[i] = (_Float16)((c < Cs) ? s[(size_t)r * Cs + c] : 0.f);
    }
}

__global__ void k_update(const float* __restrict__ w, const float* __restrict__ X,
                         float* __restrict__ wu, _Float16* __restrict__ wu16,
                         int r, int c)
{
    size_t n = (size_t)r * c;
    for (size_t i = blockIdx.x * (size_t)blockDim.x + threadIdx.x; i < n;
         i += (size_t)gridDim.x * blockDim.x) {
        int rr = (int)(i / c), cc = (int)(i % c);
        float v = TTLA * w[i] - TTLE * X[(size_t)cc * r + rr];
        wu[i] = v; wu16[i] = (_Float16)v;
    }
}

__global__ void k_residual1(const float* __restrict__ x, const float* __restrict__ attn,
                            const float* __restrict__ predu, const float* __restrict__ gate,
                            float* __restrict__ x2, size_t n)
{
    float g = 1.f / (1.f + __expf(-gate[0]));
    for (size_t i = blockIdx.x * (size_t)blockDim.x + threadIdx.x; i < n;
         i += (size_t)gridDim.x * blockDim.x)
        x2[i] = x[i] + attn[i] + g * predu[i];
}

__global__ void k_swiglu(const float* __restrict__ f1, const float* __restrict__ f3,
                         _Float16* __restrict__ o16, int R, int Cs, int ldd)
{
    size_t n = (size_t)R * ldd;
    for (size_t i = blockIdx.x * (size_t)blockDim.x + threadIdx.x; i < n;
         i += (size_t)gridDim.x * blockDim.x) {
        int r = (int)(i / ldd), c = (int)(i % ldd);
        float v = 0.f;
        if (c < Cs) {
            float u = f1[(size_t)r * Cs + c];
            v = (u / (1.f + __expf(-u))) * f3[(size_t)r * Cs + c];
        }
        o16[i] = (_Float16)v;
    }
}

__global__ void k_add2(const float* __restrict__ a, const float* __restrict__ b,
                       float* __restrict__ o, size_t n)
{
    for (size_t i = blockIdx.x * (size_t)blockDim.x + threadIdx.x; i < n;
         i += (size_t)gridDim.x * blockDim.x)
        o[i] = a[i] + b[i];
}

// =====================================================================
// Host orchestration
// =====================================================================
static inline unsigned ewb(size_t n)
{
    size_t b = (n + 255) / 256;
    return (unsigned)(b > 4096 ? 4096 : (b ? b : 1));
}

extern "C" void kernel_launch(void* const* d_in, const int* in_sizes, int n_in,
                              void* d_out, int out_size, void* d_ws, size_t ws_size,
                              hipStream_t stream)
{
    (void)in_sizes; (void)n_in; (void)out_size; (void)ws_size;

    const float* x        = (const float*)d_in[0];
    const float* norm1_w  = (const float*)d_in[1];
    const float* norm2_w  = (const float*)d_in[2];
    const float* qkv_w    = (const float*)d_in[3];
    const float* q_norm_w = (const float*)d_in[4];
    const float* k_norm_w = (const float*)d_in[5];
    const float* gamma_w1 = (const float*)d_in[6];
    const float* gamma_w2 = (const float*)d_in[7];
    const float* mem_wk   = (const float*)d_in[8];
    const float* mem_w1   = (const float*)d_in[9];
    const float* mem_w2   = (const float*)d_in[10];
    const float* mem_gate = (const float*)d_in[11];
    const float* wo_w     = (const float*)d_in[12];
    const float* ffn_w1   = (const float*)d_in[13];
    const float* ffn_w2   = (const float*)d_in[14];
    const float* ffn_w3   = (const float*)d_in[15];
    float* out = (float*)d_out;

    const size_t T    = (size_t)Bq * Sq;
    const size_t nBSD = T * Dq;

    uintptr_t cur = (uintptr_t)d_ws;
    auto allocB = [&](size_t bytes) {
        cur = (cur + 15) & ~(uintptr_t)15;
        void* r = (void*)cur; cur += bytes; return r;
    };
    auto aF = [&](size_t n) { return (float*)allocB(n * sizeof(float)); };
    auto aH = [&](size_t n) { return (_Float16*)allocB(n * sizeof(_Float16)); };

    // f32 buffers
    float* qkv     = aF(T * 3 * Dq);
    float* gammam  = aF(T * GGH);
    float* gamma   = aF(T);
    float* womega  = aF(T);
    float* Qo      = aF(nBSD);
    float* Ko      = aF(nBSD);
    float* Vflat   = aF(nBSD);
    float* z       = aF(T * HDq);
    float* u       = aF(T * MHID);
    float* pred    = aF(nBSD);
    float* partial = aF(256);
    float* scal    = aF(4);
    float* dh      = aF(T * MHID);
    float* g2      = aF((size_t)MHID * Dq);
    float* g1      = aF((size_t)FPOLY * MHID);
    float* dfeat   = aF(T * FPOLY);
    float* dz      = aF(T * HDq);
    float* gk      = aF((size_t)Dq * HDq);
    float* Xb      = aF((size_t)MHID * FPOLY);
    float* Xb2     = aF((size_t)MHID * FPOLY);
    float* Ab      = aF((size_t)MHID * MHID);
    float* A2b     = aF((size_t)MHID * MHID);
    float* wk_u    = aF((size_t)Dq * HDq);
    float* w1_u    = aF((size_t)FPOLY * MHID);
    float* w2_u    = aF((size_t)MHID * Dq);
    float* attnp   = aF(nBSD);
    float* x2      = aF(nBSD);
    float* f1      = aF(T * FFNH);
    float* f3      = aF(T * FFNH);
    float* ffn_out = aF(nBSD);

    // f16 buffers
    _Float16* h16     = aH(nBSD);
    _Float16* qkvw16  = aH((size_t)Dq * 3 * Dq);
    _Float16* gw1_16  = aH((size_t)Dq * GGH);
    _Float16* Q16     = aH(nBSD);
    _Float16* K16     = aH(nBSD);
    _Float16* Vt16    = aH(nBSD);
    _Float16* qmem16  = aH(nBSD);
    _Float16* wk16    = aH((size_t)Dq * HDq);
    _Float16* w1_16   = aH((size_t)FPOLY * MHID);
    _Float16* w2_16   = aH((size_t)MHID * Dq);
    _Float16* feat16  = aH(T * FPOLY);
    _Float16* hm16    = aH(T * MHID);
    _Float16* dpred16 = aH(nBSD);
    _Float16* dh16    = aH(T * MHID);
    _Float16* dz16    = aH(T * HDq);
    _Float16* X16     = aH((size_t)MHID * FPOLY);
    _Float16* Ab16    = aH((size_t)MHID * MHID);
    _Float16* wku16   = aH((size_t)Dq * HDq);
    _Float16* w1u16   = aH((size_t)FPOLY * MHID);
    _Float16* w2u16   = aH((size_t)MHID * Dq);
    _Float16* attn16  = aH(nBSD);
    _Float16* wo16    = aH((size_t)Dq * Dq);
    _Float16* h2_16   = aH(nBSD);
    _Float16* fw1_16  = aH((size_t)Dq * FFNHP);
    _Float16* fw3_16  = aH((size_t)Dq * FFNHP);
    _Float16* fw2_16  = aH((size_t)FFNH * Dq);
    _Float16* f1p16   = aH(T * FFNHP);

    // ---- weight conversions (f32 -> f16 scratch) ----
    k_cvtf<<<ewb((size_t)Dq*3*Dq), 256, 0, stream>>>(qkv_w, qkvw16, (size_t)Dq*3*Dq);
    k_cvtf<<<ewb((size_t)Dq*GGH), 256, 0, stream>>>(gamma_w1, gw1_16, (size_t)Dq*GGH);
    k_cvtf<<<ewb((size_t)Dq*HDq), 256, 0, stream>>>(mem_wk, wk16, (size_t)Dq*HDq);
    k_cvtf<<<ewb((size_t)FPOLY*MHID), 256, 0, stream>>>(mem_w1, w1_16, (size_t)FPOLY*MHID);
    k_cvtf<<<ewb((size_t)MHID*Dq), 256, 0, stream>>>(mem_w2, w2_16, (size_t)MHID*Dq);
    k_cvtf<<<ewb((size_t)Dq*Dq), 256, 0, stream>>>(wo_w, wo16, (size_t)Dq*Dq);
    k_cvtp<<<ewb((size_t)Dq*FFNHP), 256, 0, stream>>>(ffn_w1, fw1_16, Dq, FFNH, FFNHP);
    k_cvtp<<<ewb((size_t)Dq*FFNHP), 256, 0, stream>>>(ffn_w3, fw3_16, Dq, FFNH, FFNHP);
    k_cvtf<<<ewb((size_t)FFNH*Dq), 256, 0, stream>>>(ffn_w2, fw2_16, (size_t)FFNH*Dq);

    // ---- norm1 + projections ----
    k_rmsnorm<<<(unsigned)T, 256, 0, stream>>>(x, norm1_w, h16);
    gemm(stream, h16, qkvw16, qkv, (int)T, 3*Dq, Dq, Dq, 3*Dq, 3*Dq, 0, 0);
    gemm(stream, h16, gw1_16, gammam, (int)T, GGH, Dq, Dq, GGH, GGH, 0, 0, 1);
    k_gamma<<<(unsigned)T, 64, 0, stream>>>(gammam, gamma_w2, gamma, womega);
    k_qkv_heads<<<(unsigned)(T * Hq), 64, 0, stream>>>(qkv, q_norm_w, k_norm_w,
                                                       Qo, Ko, Q16, K16, Vt16, Vflat);
    k_kcm_qmem<<<Bq * Hq, 64, 0, stream>>>(Ko, Qo, gamma, qmem16);

    // ---- memory forward (original weights) ----
    gemm(stream, qmem16, wk16, z, (int)T, HDq, Dq, Dq, HDq, HDq, 0, 0);
    k_phi2<<<(unsigned)T, 256, 0, stream>>>(z, feat16);
    gemm(stream, feat16, w1_16, u, (int)T, MHID, FPOLY, FPOLY, MHID, MHID, 0, 0);
    k_silu_h<<<ewb(T*MHID), 256, 0, stream>>>(u, hm16, T*MHID);
    gemm(stream, hm16, w2_16, pred, (int)T, Dq, MHID, MHID, Dq, Dq, 0, 0);

    // ---- gradients of the omega loss ----
    k_reduce_partial<<<256, 256, 0, stream>>>(womega, T, partial, 0);
    k_reduce_final<<<1, 256, 0, stream>>>(partial, 256, scal + 0);
    k_dpred<<<ewb(nBSD), 256, 0, stream>>>(pred, Vflat, womega, scal + 0, dpred16, nBSD);
    gemm(stream, dpred16, w2_16, dh, (int)T, MHID, Dq, Dq, Dq, MHID, 0, 1);  // dpred @ w2^T
    k_silu_bwd<<<ewb(T*MHID), 256, 0, stream>>>(dh, u, dh16, T*MHID);        // du (f16)
    gemm(stream, hm16, dpred16, g2, MHID, Dq, (int)T, MHID, Dq, Dq, 1, 0);   // hm^T @ dpred
    gemm(stream, feat16, dh16, g1, FPOLY, MHID, (int)T, FPOLY, MHID, MHID, 1, 0); // feat^T @ du
    gemm(stream, dh16, w1_16, dfeat, (int)T, FPOLY, MHID, MHID, MHID, FPOLY, 0, 1); // du @ w1^T
    k_dz<<<(unsigned)T, 64, 0, stream>>>(dfeat, z, dz);
    k_cvtf<<<ewb(T*HDq), 256, 0, stream>>>(dz, dz16, T*HDq);
    gemm(stream, qmem16, dz16, gk, Dq, HDq, (int)T, Dq, HDq, HDq, 1, 0);     // qmem^T @ dz

    // ---- Newton-Schulz + weight update (all grads tall: X = G^T/||G||) ----
    auto newton_update = [&](const float* G, int r, int c, const float* w,
                             float* wu, _Float16* wu16) {
        k_reduce_partial<<<256, 256, 0, stream>>>(G, (size_t)r*c, partial, 1);
        k_reduce_final<<<1, 256, 0, stream>>>(partial, 256, scal + 1);
        k_transpose_scale<<<ewb((size_t)r*c), 256, 0, stream>>>(G, Xb, r, c, scal + 1);
        float* X = Xb; float* Xn = Xb2;
        for (int it = 0; it < 5; ++it) {
            k_cvtf<<<ewb((size_t)r*c), 256, 0, stream>>>(X, X16, (size_t)r*c);
            gemm(stream, X16, X16, Ab, c, c, r, r, r, c, 0, 1);      // A = X X^T
            k_cvtf<<<ewb((size_t)c*c), 256, 0, stream>>>(Ab, Ab16, (size_t)c*c);
            gemm(stream, Ab16, Ab16, A2b, c, c, c, c, c, c, 0, 0);   // A2 = A A
            k_ns_combine<<<ewb((size_t)c*c), 256, 0, stream>>>(Ab, A2b, Ab, (size_t)c*c);
            k_cvtf<<<ewb((size_t)c*c), 256, 0, stream>>>(Ab, Ab16, (size_t)c*c); // T (f16)
            gemm(stream, Ab16, X16, Xn, c, r, c, c, r, r, 0, 0, 0, X, NSA);      // Xn = T@X + a*X
            float* t = X; X = Xn; Xn = t;
        }
        k_update<<<ewb((size_t)r*c), 256, 0, stream>>>(w, X, wu, wu16, r, c);
    };
    newton_update(gk, Dq, HDq, mem_wk, wk_u, wku16);
    newton_update(g1, FPOLY, MHID, mem_w1, w1_u, w1u16);
    newton_update(g2, MHID, Dq, mem_w2, w2_u, w2u16);

    // ---- memory forward with updated weights ----
    gemm(stream, qmem16, wku16, z, (int)T, HDq, Dq, Dq, HDq, HDq, 0, 0);
    k_phi2<<<(unsigned)T, 256, 0, stream>>>(z, feat16);
    gemm(stream, feat16, w1u16, u, (int)T, MHID, FPOLY, FPOLY, MHID, MHID, 0, 0);
    k_silu_h<<<ewb(T*MHID), 256, 0, stream>>>(u, hm16, T*MHID);
    gemm(stream, hm16, w2u16, pred, (int)T, Dq, MHID, MHID, Dq, Dq, 0, 0);

    // ---- attention ----
    k_flash<<<Bq * Hq * (Sq / 64), 128, 0, stream>>>(Q16, K16, Vt16, attn16);
    gemm(stream, attn16, wo16, attnp, (int)T, Dq, Dq, Dq, Dq, Dq, 0, 0);

    // ---- residual 1 + FFN ----
    k_residual1<<<ewb(nBSD), 256, 0, stream>>>(x, attnp, pred, mem_gate, x2, nBSD);
    k_rmsnorm<<<(unsigned)T, 256, 0, stream>>>(x2, norm2_w, h2_16);
    gemm(stream, h2_16, fw1_16, f1, (int)T, FFNH, Dq, Dq, FFNHP, FFNH, 0, 0);
    gemm(stream, h2_16, fw3_16, f3, (int)T, FFNH, Dq, Dq, FFNHP, FFNH, 0, 0);
    k_swiglu<<<ewb(T*FFNHP), 256, 0, stream>>>(f1, f3, f1p16, (int)T, FFNH, FFNHP);
    gemm(stream, f1p16, fw2_16, ffn_out, (int)T, Dq, FFNH, FFNHP, Dq, Dq, 0, 0);
    k_add2<<<ewb(nBSD), 256, 0, stream>>>(x2, ffn_out, out, nBSD);
}
